// V2TransfuserModelARPlus_37151467111317
// MI455X (gfx1250) — matrix operand order repaired and checked by reading, not compiled, not run on hardware
//
#include <hip/hip_runtime.h>
#include <hip/hip_bf16.h>
#include <math.h>

// ---------------------------------------------------------------------------
// V2TransfuserModelARPlus for MI455X (gfx1250, wave32, WMMA).
// - All dense projections: bf16 WMMA GEMM (v_wmma_f32_16x16x32_bf16, f32 acc).
// - bf16-A GEMM (gated-kv path, the FLOP-dominant projection) stages its A tile
//   with GLOBAL_LOAD_ASYNC_TO_LDS_B128 + s_wait_asynccnt (CDNA5 async path).
// - BEV cross-attention (P=2048): WMMA flash attention computing S^T = K.Q^T so
//   the S^T C-layout registers ARE the A-layout P registers for O += P.V.
// - Small reductions (T=8, K=32 attn, LN, gating): wave32 VALU kernels.
// - kv-path intermediates stored bf16 -> whole working set < 192 MB L2.
// ---------------------------------------------------------------------------

typedef __bf16 bf16;
typedef __attribute__((ext_vector_type(16))) __bf16 bf16x16;
typedef __attribute__((ext_vector_type(8)))  __bf16 bf16x8;
typedef __attribute__((ext_vector_type(4)))  __bf16 bf16x4;
typedef __attribute__((ext_vector_type(8)))  float  f32x8;

#define B_  8
#define M_  20
#define T_  8
#define D_  256
#define K_  32
#define P_  2048
#define L_  3
#define H_  8
#define F_  1024
#define NTOK (B_*M_*T_)      /* 1280  */
#define NAGT (B_*K_*T_)      /* 2048  */
#define NKV  (B_*M_*K_*T_)   /* 40960 */
#define NBEV (B_*P_)         /* 16384 */

#define SCALE_ATTN 0.17677669529663687f   /* 1/sqrt(32) */

__constant__ float kScale = SCALE_ATTN;

// ---------------------------------------------------------------------------
// helpers
// ---------------------------------------------------------------------------
__device__ __forceinline__ bf16x4 ld4_as_bf16(const float* p) {
  float4 v = *(const float4*)p;
  bf16x4 r; r[0] = (bf16)v.x; r[1] = (bf16)v.y; r[2] = (bf16)v.z; r[3] = (bf16)v.w;
  return r;
}
__device__ __forceinline__ bf16x4 ld4_as_bf16(const bf16* p) {
  return *(const bf16x4*)p;
}
__device__ __forceinline__ f32x8 wmma_bf16(bf16x16 a, bf16x16 b, f32x8 c) {
  return __builtin_amdgcn_wmma_f32_16x16x32_bf16(false, a, false, b,
                                                 (short)0, c, false, false);
}

// ---------------------------------------------------------------------------
// Tiled WMMA GEMM:  C[M,N] = A[M,K] @ W[N,K]^T + bias[N]   (optional ReLU)
// Workgroup: 256 threads = 8 waves, 64(M) x 128(N) tile, 32x32 per wave.
// Requires: M % 64 == 0, N % 128 == 0, K % 32 == 0 (true for all calls here).
// ---------------------------------------------------------------------------
#define LSTR 40   // LDS row stride in bf16 elems (80B: 16B-aligned, bank-spread)

template<typename AT, typename OT, bool RELU>
__global__ __launch_bounds__(256)
void gemm_wmma(const AT* __restrict__ A, const float* __restrict__ W,
               const float* __restrict__ bias, OT* __restrict__ C,
               int M, int N, int K)
{
  __shared__ bf16 lA[64][LSTR];
  __shared__ bf16 lB[128][LSTR];

  const int tid  = threadIdx.x;
  const int lane = tid & 31;
  const int wave = tid >> 5;       // 0..7
  const int wm   = wave & 1;       // 2 row-groups of 32
  const int wn   = wave >> 1;      // 4 col-groups of 32
  const int m0   = blockIdx.x * 64;
  const int n0   = blockIdx.y * 128;
  const int r    = lane & 15;      // row/col within 16-tile
  const int kh   = lane >> 4;      // K-half select per ISA layout

  f32x8 acc[2][2];
  for (int i = 0; i < 2; ++i)
    for (int j = 0; j < 2; ++j)
      for (int e = 0; e < 8; ++e) acc[i][j][e] = 0.0f;

  for (int k0 = 0; k0 < K; k0 += 32) {
    // ---- stage A tile (64x32) as bf16 ----
    if constexpr (__is_same(AT, bf16)) {
      // A is already bf16 in memory: raw async global->LDS copy, one B128
      // per lane (64 rows x 64B = 256 x 16B), tracked by ASYNCcnt.
      int row = tid >> 2;
      int seg = tid & 3;                 // 16-byte segment within the row
      const bf16* gp = A + (size_t)(m0 + row) * K + k0 + seg * 8;
      unsigned lds_addr = (unsigned)(size_t)&lA[row][seg * 8];
      asm volatile("global_load_async_to_lds_b128 %0, %1, off"
                   :: "v"(lds_addr), "v"(gp) : "memory");
    } else {
      #pragma unroll
      for (int it = 0; it < 2; ++it) {
        int idx = tid + it * 256;          // 512 x float4-slots
        int row = idx >> 3;
        int c4  = idx & 7;
        bf16x4 v = ld4_as_bf16(A + (size_t)(m0 + row) * K + k0 + c4 * 4);
        *(bf16x4*)&lA[row][c4 * 4] = v;
      }
    }
    // ---- stage W tile (128x32) as bf16 (f32 -> bf16 convert in VGPRs) ----
    #pragma unroll
    for (int it = 0; it < 4; ++it) {
      int idx = tid + it * 256;            // 1024 x float4-slots
      int row = idx >> 3;
      int c4  = idx & 7;
      bf16x4 v = ld4_as_bf16(W + (size_t)(n0 + row) * K + k0 + c4 * 4);
      *(bf16x4*)&lB[row][c4 * 4] = v;
    }
    // hint next K-tiles into cache (global_prefetch_b8)
    if (k0 + 32 < K) {
      int row = tid >> 3, c4 = tid & 7;
      __builtin_prefetch(A + (size_t)(m0 + row) * K + (k0 + 32) + c4 * 4, 0, 1);
      __builtin_prefetch(W + (size_t)(n0 + row) * K + (k0 + 32) + c4 * 4, 0, 1);
    }
    if constexpr (__is_same(AT, bf16)) {
      asm volatile("s_wait_asynccnt 0x0" ::: "memory");
    }
    __syncthreads();

    // ---- build fragments (ISA 7.12.2 layouts) ----
    bf16x16 afrag[2], bfrag[2];
    #pragma unroll
    for (int f = 0; f < 2; ++f) {
      // A 16x32: lane half kh -> elems 0..7 = K[kh*8..], elems 8..15 = K[16+kh*8..]
      int arow = wm * 32 + f * 16 + r;
      bf16x8 alo = *(const bf16x8*)&lA[arow][kh * 8];
      bf16x8 ahi = *(const bf16x8*)&lA[arow][16 + kh * 8];
      afrag[f] = __builtin_shufflevector(alo, ahi,
                   0,1,2,3,4,5,6,7,8,9,10,11,12,13,14,15);
      // B 32x16: lane half kh -> elems 0..7 = K[kh*8..], elems 8..15 = K[16+kh*8..]
      int bcol = wn * 32 + f * 16 + r;
      bf16x8 blo = *(const bf16x8*)&lB[bcol][kh * 8];
      bf16x8 bhi = *(const bf16x8*)&lB[bcol][16 + kh * 8];
      bfrag[f] = __builtin_shufflevector(blo, bhi,
                   0,1,2,3,4,5,6,7,8,9,10,11,12,13,14,15);
    }

    #pragma unroll
    for (int fm = 0; fm < 2; ++fm)
      #pragma unroll
      for (int fn = 0; fn < 2; ++fn)
        acc[fm][fn] = wmma_bf16(afrag[fm], bfrag[fn], acc[fm][fn]);

    __syncthreads();
  }

  // ---- epilogue: C/D layout = lane(0..15)->N, +kh*8 rows, vgpr->row ----
  #pragma unroll
  for (int fm = 0; fm < 2; ++fm)
    #pragma unroll
    for (int fn = 0; fn < 2; ++fn) {
      int n   = n0 + wn * 32 + fn * 16 + r;
      float bv = bias ? bias[n] : 0.0f;
      int mb  = m0 + wm * 32 + fm * 16 + kh * 8;
      #pragma unroll
      for (int rr = 0; rr < 8; ++rr) {
        float v = acc[fm][fn][rr] + bv;
        if (RELU) v = fmaxf(v, 0.0f);
        C[(size_t)(mb + rr) * N + n] = (OT)v;
      }
    }
}

// ---------------------------------------------------------------------------
// Causal temporal self-attention over T=8.  One block (64 thr) per (b,m);
// thread = (h, t).  qkv layout [NTOK][768] = [q|k|v], head stride 32.
// ---------------------------------------------------------------------------
__global__ __launch_bounds__(64)
void temporal_attn(const float* __restrict__ qkv, float* __restrict__ o)
{
  int bm = blockIdx.x;
  int h  = threadIdx.x >> 3;
  int t  = threadIdx.x & 7;

  const float* qp = qkv + (size_t)(bm * T_ + t) * 768 + h * 32;
  float q[32];
  #pragma unroll
  for (int d = 0; d < 32; ++d) q[d] = qp[d];

  float s[T_];
  float mx = -__builtin_inff();
  for (int j = 0; j <= t; ++j) {
    const float* kp = qkv + (size_t)(bm * T_ + j) * 768 + 256 + h * 32;
    float a = 0.0f;
    #pragma unroll
    for (int d = 0; d < 32; ++d) a += q[d] * kp[d];
    s[j] = a * kScale;
    mx = fmaxf(mx, s[j]);
  }
  float sum = 0.0f;
  for (int j = 0; j <= t; ++j) { s[j] = __expf(s[j] - mx); sum += s[j]; }
  float inv = 1.0f / sum;

  float out[32];
  #pragma unroll
  for (int d = 0; d < 32; ++d) out[d] = 0.0f;
  for (int j = 0; j <= t; ++j) {
    const float* vp = qkv + (size_t)(bm * T_ + j) * 768 + 512 + h * 32;
    float p = s[j] * inv;
    #pragma unroll
    for (int d = 0; d < 32; ++d) out[d] += p * vp[d];
  }
  float* op = o + (size_t)(bm * T_ + t) * D_ + h * 32;
  #pragma unroll
  for (int d = 0; d < 32; ++d) op[d] = out[d];
}

// ---------------------------------------------------------------------------
// x = LayerNorm(x + oproj) * g + b    (one wave per token, wave32 reductions)
// ---------------------------------------------------------------------------
__global__ __launch_bounds__(256)
void add_ln(float* __restrict__ x, const float* __restrict__ oproj,
            const float* __restrict__ gain, const float* __restrict__ beta,
            int ntok)
{
  int wave = threadIdx.x >> 5, lane = threadIdx.x & 31;
  int tok = blockIdx.x * 8 + wave;
  if (tok >= ntok) return;

  float v[8];
  float s = 0.0f;
  #pragma unroll
  for (int j = 0; j < 8; ++j) {
    int c = lane + j * 32;
    v[j] = x[(size_t)tok * D_ + c] + oproj[(size_t)tok * D_ + c];
    s += v[j];
  }
  #pragma unroll
  for (int off = 16; off; off >>= 1) s += __shfl_xor(s, off, 32);
  float mu = s * (1.0f / 256.0f);

  float var = 0.0f;
  #pragma unroll
  for (int j = 0; j < 8; ++j) { float d = v[j] - mu; var += d * d; }
  #pragma unroll
  for (int off = 16; off; off >>= 1) var += __shfl_xor(var, off, 32);
  var *= (1.0f / 256.0f);
  float rstd = rsqrtf(var + 1e-5f);

  #pragma unroll
  for (int j = 0; j < 8; ++j) {
    int c = lane + j * 32;
    x[(size_t)tok * D_ + c] = (v[j] - mu) * rstd * gain[c] + beta[c];
  }
}

// ---------------------------------------------------------------------------
// additive key mask: 0 if valid or the whole row is invalid, else -inf.
// One wave per batch (K=32 = lane).
// ---------------------------------------------------------------------------
__global__ __launch_bounds__(256)
void make_mask(const int* __restrict__ valid, float* __restrict__ maskadd)
{
  int b = threadIdx.x >> 5, k = threadIdx.x & 31;
  int vi  = valid[b * K_ + k];
  int tot = (vi != 0) ? 1 : 0;
  #pragma unroll
  for (int off = 16; off; off >>= 1) tot += __shfl_xor(tot, off, 32);
  maskadd[b * K_ + k] = (tot == 0 || vi != 0) ? 0.0f : -__builtin_inff();
}

// ---------------------------------------------------------------------------
// gate = sigmoid(tanh(gq + gk) . g_Wo + g_bo);  kv = bf16(agent_kv * gate)
// One wave per (b,m,k,t).
// ---------------------------------------------------------------------------
__global__ __launch_bounds__(256)
void gate_kv(const float* __restrict__ gq, const float* __restrict__ gk,
             const float* __restrict__ agent_kv, const float* __restrict__ gWo,
             const float* __restrict__ gbo, bf16* __restrict__ kvb)
{
  int wave = threadIdx.x >> 5, lane = threadIdx.x & 31;
  long idx = (long)blockIdx.x * 8 + wave;          // ((b*M+m)*K + k)*T + t
  int t = idx & 7;
  int k = (idx >> 3) & 31;
  long bm = idx >> 8;                              // K_*T_ == 256
  int b = (int)(bm / M_);
  long bmt = bm * T_ + t;
  long bkt = ((long)(b * K_ + k)) * T_ + t;

  const float* gqp = gq + bmt * D_;
  const float* gkp = gk + bkt * D_;
  float sum = 0.0f;
  #pragma unroll
  for (int j = 0; j < 8; ++j) {
    int c = lane + j * 32;
    sum += tanhf(gqp[c] + gkp[c]) * gWo[c];
  }
  #pragma unroll
  for (int off = 16; off; off >>= 1) sum += __shfl_xor(sum, off, 32);
  float gate = 1.0f / (1.0f + __expf(-(sum + gbo[0])));

  const float* av = agent_kv + bkt * D_;
  bf16* out = kvb + idx * D_;
  #pragma unroll
  for (int j = 0; j < 8; ++j) {
    int c = lane + j * 32;
    out[c] = (bf16)(av[c] * gate);
  }
}

// ---------------------------------------------------------------------------
// ego->agent cross-attention over K=32 agents (masked softmax).
// One thread per (b,m,t,h).  ekv layout [NKV][512] = [e_k | e_v], bf16.
// ---------------------------------------------------------------------------
__global__ __launch_bounds__(256)
void e_attn(const float* __restrict__ eq, const bf16* __restrict__ ekv,
            const float* __restrict__ maskadd, float* __restrict__ o)
{
  long i = (long)blockIdx.x * 256 + threadIdx.x;   // B*M*T*H = 10240
  int h = i & 7;
  int t = (i >> 3) & 7;
  long bm = i >> 6;
  int b = (int)(bm / M_);
  long bmt = bm * T_ + t;

  const float* qp = eq + bmt * D_ + h * 32;
  float q[32];
  #pragma unroll
  for (int d = 0; d < 32; ++d) q[d] = qp[d];

  float s[K_];
  float mx = -__builtin_inff();
  for (int k = 0; k < K_; ++k) {
    long row = (bm * K_ + k) * T_ + t;
    const bf16* kp = ekv + row * 512 + h * 32;
    float a = 0.0f;
    #pragma unroll
    for (int d = 0; d < 32; ++d) a += q[d] * (float)kp[d];
    a = a * kScale + maskadd[b * K_ + k];
    s[k] = a;
    mx = fmaxf(mx, a);
  }
  float sum = 0.0f;
  for (int k = 0; k < K_; ++k) { s[k] = __expf(s[k] - mx); sum += s[k]; }
  float inv = 1.0f / sum;

  float out[32];
  #pragma unroll
  for (int d = 0; d < 32; ++d) out[d] = 0.0f;
  for (int k = 0; k < K_; ++k) {
    long row = (bm * K_ + k) * T_ + t;
    const bf16* vp = ekv + row * 512 + 256 + h * 32;
    float p = s[k] * inv;
    #pragma unroll
    for (int d = 0; d < 32; ++d) out[d] += p * (float)vp[d];
  }
  float* op = o + bmt * D_ + h * 32;
  #pragma unroll
  for (int d = 0; d < 32; ++d) op[d] = out[d];
}

// ---------------------------------------------------------------------------
// BEV cross-attention, WMMA flash-attention form.  2 waves per block; each
// wave owns one (b, h, 16-query tile); 10 tiles cover a batch's 160 queries.
//
// Per 32-key step:
//   S^T[32k,16q] = K.(scaled Q)^T        -> 2x v_wmma (C-layout: lane=q, vgpr=key)
//   online softmax: 1 shfl_xor(16) max, 1 shfl_xor(16) sum, 8 bpermute rescale
//   P(A-layout) == S^T(C-layout) element-for-element -> in-lane bf16 convert
//   O[16q,32d] += P.V                    -> 2x v_wmma (V via coalesced gather)
// ---------------------------------------------------------------------------
__global__ __launch_bounds__(64)
void bev_attn_wmma(const float* __restrict__ bq, const bf16* __restrict__ bkv,
                   float* __restrict__ o)
{
  const int lane = threadIdx.x & 31;
  const int wv   = threadIdx.x >> 5;   // 0..1
  const int n    = lane & 15;      // q column / dh column
  const int kh   = lane >> 4;      // half select
  int blk = blockIdx.x;            // B*H*5 = 320
  int b   = blk / (H_ * 5);
  int rem = blk % (H_ * 5);
  int h   = rem / 5;
  int qt  = (rem % 5) * 2 + wv;    // 0..9

  // Q^T B-fragment: lane = q col, elems = dh[kh*16 .. kh*16+15], scale folded in
  const float* qrow = bq + (size_t)(b * 160 + qt * 16 + n) * D_ + h * 32 + kh * 16;
  bf16x16 qfrag;
  #pragma unroll
  for (int e = 0; e < 16; ++e) qfrag[e] = (bf16)(qrow[e] * SCALE_ATTN);

  f32x8 o0, o1;
  #pragma unroll
  for (int e = 0; e < 8; ++e) { o0[e] = 0.0f; o1[e] = 0.0f; }
  float mx = -__builtin_inff();
  float l  = 0.0f;

  const bf16* kbase = bkv + (size_t)b * P_ * 512 + h * 32;
  const bf16* vbase = bkv + (size_t)b * P_ * 512 + 256 + h * 32;

  for (int p0 = 0; p0 < P_; p0 += 32) {
    // K A-fragments for key rows [p0..p0+15] and [p0+16..p0+31]
    bf16x16 kf[2];
    #pragma unroll
    for (int s = 0; s < 2; ++s) {
      const bf16* kr = kbase + (size_t)(p0 + s * 16 + n) * 512;
      bf16x8 lo = *(const bf16x8*)(kr + kh * 16);
      bf16x8 hi = *(const bf16x8*)(kr + kh * 16 + 8);
      kf[s] = __builtin_shufflevector(lo, hi,
                0,1,2,3,4,5,6,7,8,9,10,11,12,13,14,15);
    }
    f32x8 z;
    #pragma unroll
    for (int e = 0; e < 8; ++e) z[e] = 0.0f;
    f32x8 st0 = wmma_bf16(kf[0], qfrag, z);   // keys p0+0..15   x q
    f32x8 st1 = wmma_bf16(kf[1], qfrag, z);   // keys p0+16..31  x q

    // --- online softmax over this step's 32 keys (per q column) ---
    float smax = st0[0];
    #pragma unroll
    for (int e = 0; e < 8; ++e) smax = fmaxf(smax, fmaxf(st0[e], st1[e]));
    smax = fmaxf(smax, __shfl_xor(smax, 16, 32));
    float nm   = fmaxf(mx, smax);
    float corr = __expf(mx - nm);
    mx = nm;

    float ssum = 0.0f;
    #pragma unroll
    for (int e = 0; e < 8; ++e) {
      st0[e] = __expf(st0[e] - nm);
      st1[e] = __expf(st1[e] - nm);
      ssum += st0[e] + st1[e];
    }
    ssum += __shfl_xor(ssum, 16, 32);
    l = l * corr + ssum;

    // rescale O accumulators (row q = rr + 8*kh lives in lane q of corr)
    #pragma unroll
    for (int rr = 0; rr < 8; ++rr) {
      float c = __shfl(corr, rr + kh * 8, 32);
      o0[rr] *= c;
      o1[rr] *= c;
    }

    // P A-fragment: identical register mapping to S^T C-layout -> just convert
    bf16x16 pf;
    #pragma unroll
    for (int e = 0; e < 8; ++e) {
      pf[e]     = (bf16)st0[e];
      pf[8 + e] = (bf16)st1[e];
    }

    // V B-fragments: lane = dh col, elems walk 16 key rows (coalesced gather)
    bf16x16 v0, v1;
    const bf16* vc0 = vbase + (size_t)(p0 + kh * 8) * 512 + n;
    const bf16* vc1 = vc0 + 16;
    #pragma unroll
    for (int e = 0; e < 16; ++e) {
      v0[e] = vc0[(size_t)(e + (e >= 8 ? 8 : 0)) * 512];
      v1[e] = vc1[(size_t)(e + (e >= 8 ? 8 : 0)) * 512];
    }
    o0 = wmma_bf16(pf, v0, o0);   // dh 0..15
    o1 = wmma_bf16(pf, v1, o1);   // dh 16..31
  }

  float linv = 1.0f / l;
  #pragma unroll
  for (int rr = 0; rr < 8; ++rr) {
    float li = __shfl(linv, rr + kh * 8, 32);
    size_t row = (size_t)b * 160 + qt * 16 + rr + kh * 8;
    float* op = o + row * D_ + h * 32;
    op[n]      = o0[rr] * li;
    op[16 + n] = o1[rr] * li;
  }
}

// ---------------------------------------------------------------------------
// host launcher
// ---------------------------------------------------------------------------
extern "C" void kernel_launch(void* const* d_in, const int* in_sizes, int n_in,
                              void* d_out, int out_size, void* d_ws, size_t ws_size,
                              hipStream_t stream)
{
  (void)in_sizes; (void)n_in; (void)out_size; (void)ws_size;

  const float* ego_q    = (const float*)d_in[0];
  const float* agent_kv = (const float*)d_in[1];
  const float* bev_feat = (const float*)d_in[2];
  const int*   topk_val = (const int*)  d_in[3];
  const float* t_Wqkv = (const float*)d_in[4];  const float* t_bqkv = (const float*)d_in[5];
  const float* t_Wo   = (const float*)d_in[6];  const float* t_bo   = (const float*)d_in[7];
  const float* t_ng   = (const float*)d_in[8];  const float* t_nb   = (const float*)d_in[9];
  const float* e_Wqkv = (const float*)d_in[10]; const float* e_bqkv = (const float*)d_in[11];
  const float* e_Wo   = (const float*)d_in[12]; const float* e_bo   = (const float*)d_in[13];
  const float* e_ng   = (const float*)d_in[14]; const float* e_nb   = (const float*)d_in[15];
  const float* b_Wqkv = (const float*)d_in[16]; const float* b_bqkv = (const float*)d_in[17];
  const float* b_Wo   = (const float*)d_in[18]; const float* b_bo   = (const float*)d_in[19];
  const float* b_ng   = (const float*)d_in[20]; const float* b_nb   = (const float*)d_in[21];
  const float* f_W1   = (const float*)d_in[22]; const float* f_b1   = (const float*)d_in[23];
  const float* f_W2   = (const float*)d_in[24]; const float* f_b2   = (const float*)d_in[25];
  const float* f_ng   = (const float*)d_in[26]; const float* f_nb   = (const float*)d_in[27];
  const float* g_Wq   = (const float*)d_in[28]; const float* g_bq   = (const float*)d_in[29];
  const float* g_Wk   = (const float*)d_in[30]; const float* g_bk   = (const float*)d_in[31];
  const float* g_Wo   = (const float*)d_in[32]; const float* g_bo   = (const float*)d_in[33];

  // ---- carve workspace (everything < ~100 MB -> resident in 192 MB L2) ----
  char*  wp  = (char*)d_ws;
  size_t off = 0;
  auto carve = [&](size_t bytes) -> void* {
    void* p = wp + off;
    off += (bytes + 255) & ~(size_t)255;
    return p;
  };
  float* x     = (float*)carve((size_t)NTOK * D_ * 4);
  float* qkv   = (float*)carve((size_t)NTOK * 768 * 4);
  float* obuf  = (float*)carve((size_t)NTOK * D_ * 4);
  float* o2    = (float*)carve((size_t)NTOK * D_ * 4);
  float* gq    = (float*)carve((size_t)NTOK * D_ * 4);
  float* gk    = (float*)carve((size_t)NAGT * D_ * 4);
  float* eq    = (float*)carve((size_t)NTOK * D_ * 4);
  float* bq    = (float*)carve((size_t)NTOK * D_ * 4);
  float* mask  = (float*)carve((size_t)B_ * K_ * 4);
  float* ffnh  = (float*)carve((size_t)NTOK * F_ * 4);
  bf16*  kvb   = (bf16*)carve((size_t)NKV  * D_ * 2);
  bf16*  ekvb  = (bf16*)carve((size_t)NKV  * 512 * 2);
  bf16*  bkvb  = (bf16*)carve((size_t)NBEV * 512 * 2);

  const dim3 blk(256);

  // x <- ego_q
  hipMemcpyAsync(x, ego_q, (size_t)NTOK * D_ * 4, hipMemcpyDeviceToDevice, stream);
  // key-padding mask (layer-invariant)
  make_mask<<<1, blk, 0, stream>>>(topk_val, mask);
  // gk = agent_kv @ g_Wk^T + g_bk (layer-invariant)
  gemm_wmma<float, float, false><<<dim3(NAGT/64, 256/128), blk, 0, stream>>>(
      agent_kv, g_Wk, g_bk, gk, NAGT, 256, 256);

  for (int i = 0; i < L_; ++i) {
    const float* tW = t_Wqkv + (size_t)i * 768 * 256;
    const float* eW = e_Wqkv + (size_t)i * 768 * 256;
    const float* bW = b_Wqkv + (size_t)i * 768 * 256;

    // --- temporal self-attention ---
    gemm_wmma<float, float, false><<<dim3(NTOK/64, 768/128), blk, 0, stream>>>(
        x, tW, t_bqkv + (size_t)i * 768, qkv, NTOK, 768, 256);
    temporal_attn<<<B_ * M_, dim3(64), 0, stream>>>(qkv, obuf);
    gemm_wmma<float, float, false><<<dim3(NTOK/64, 256/128), blk, 0, stream>>>(
        obuf, t_Wo + (size_t)i * 256 * 256, t_bo + (size_t)i * 256, o2, NTOK, 256, 256);
    add_ln<<<NTOK/8, blk, 0, stream>>>(x, o2, t_ng + (size_t)i * 256, t_nb + (size_t)i * 256, NTOK);

    // --- gated ego->agent cross-attention ---
    gemm_wmma<float, float, false><<<dim3(NTOK/64, 256/128), blk, 0, stream>>>(
        x, g_Wq, g_bq, gq, NTOK, 256, 256);
    gate_kv<<<NKV/8, blk, 0, stream>>>(gq, gk, agent_kv, g_Wo, g_bo, kvb);
    gemm_wmma<float, float, false><<<dim3(NTOK/64, 256/128), blk, 0, stream>>>(
        x, eW, e_bqkv + (size_t)i * 768, eq, NTOK, 256, 256);
    gemm_wmma<bf16, bf16, false><<<dim3(NKV/64, 512/128), blk, 0, stream>>>(
        kvb, eW + 256 * 256, e_bqkv + (size_t)i * 768 + 256, ekvb, NKV, 512, 256);
    e_attn<<<(B_*M_*T_*H_)/256, blk, 0, stream>>>(eq, ekvb, mask, obuf);
    gemm_wmma<float, float, false><<<dim3(NTOK/64, 256/128), blk, 0, stream>>>(
        obuf, e_Wo + (size_t)i * 256 * 256, e_bo + (size_t)i * 256, o2, NTOK, 256, 256);
    add_ln<<<NTOK/8, blk, 0, stream>>>(x, o2, e_ng + (size_t)i * 256, e_nb + (size_t)i * 256, NTOK);

    // --- BEV cross-attention (WMMA flash attention) ---
    gemm_wmma<float, float, false><<<dim3(NTOK/64, 256/128), blk, 0, stream>>>(
        x, bW, b_bqkv + (size_t)i * 768, bq, NTOK, 256, 256);
    gemm_wmma<float, bf16, false><<<dim3(NBEV/64, 512/128), blk, 0, stream>>>(
        bev_feat, bW + 256 * 256, b_bqkv + (size_t)i * 768 + 256, bkvb, NBEV, 512, 256);
    bev_attn_wmma<<<B_ * H_ * 5, dim3(64), 0, stream>>>(bq, bkvb, obuf);
    gemm_wmma<float, float, false><<<dim3(NTOK/64, 256/128), blk, 0, stream>>>(
        obuf, b_Wo + (size_t)i * 256 * 256, b_bo + (size_t)i * 256, o2, NTOK, 256, 256);
    add_ln<<<NTOK/8, blk, 0, stream>>>(x, o2, b_ng + (size_t)i * 256, b_nb + (size_t)i * 256, NTOK);

    // --- FFN ---
    gemm_wmma<float, float, true><<<dim3(NTOK/64, F_/128), blk, 0, stream>>>(
        x, f_W1 + (size_t)i * F_ * 256, f_b1 + (size_t)i * F_, ffnh, NTOK, F_, 256);
    gemm_wmma<float, float, false><<<dim3(NTOK/64, 256/128), blk, 0, stream>>>(
        ffnh, f_W2 + (size_t)i * 256 * F_, f_b2 + (size_t)i * 256, o2, NTOK, 256, F_);
    add_ln<<<NTOK/8, blk, 0, stream>>>(x, o2, f_ng + (size_t)i * 256, f_nb + (size_t)i * 256, NTOK);
  }

  hipMemcpyAsync(d_out, x, (size_t)NTOK * D_ * 4, hipMemcpyDeviceToDevice, stream);
}
